// MultiConvAttn_16655883173924
// MI455X (gfx1250) — matrix operand order repaired and checked by reading, not compiled, not run on hardware
//
#include <hip/hip_runtime.h>

// ---------------------------------------------------------------------------
// MultiConvAttn on gfx1250 (MI455X), wave32 + v_wmma_f32_16x16x32_bf16.
//
//   Phase 1: A[ok=o*32+kw][c] = Q[o,c,:]·W_q[:,kw] + b_q[kw], stored as
//            bf16 hi/lo planes (3-term bf16 split ~= fp32 precision).
//   Phase 2: per n: S[ok,t] = sum_c A[ok,c]*key_seq[n,t,c]  (NT GEMM,
//            M=128, K=1024, N=8192) via WMMA, keys split to bf16 hi/lo in LDS.
//   Phase 3: out[n,o,l] = sum_{k<32} S[n, o*32+k, l+k-16]  (bounds-checked).
// ---------------------------------------------------------------------------

constexpr int BN  = 4;      // batch
constexpr int SQ  = 1024;   // query length (== D)
constexpr int SK  = 8192;   // key length
constexpr int DM  = 1024;   // d_model
constexpr int KW  = 32;     // conv kernel width (H*K)
constexpr int OK  = BN * KW;        // 128 GEMM rows
constexpr int PAD = 16;
constexpr int LOUT = SK + 2 * PAD - KW + 1;   // 8193

constexpr int NT_BLK = 128;  // t-rows per block (phase 2)
constexpr int KSTEP  = 32;   // K step per WMMA round
constexpr int BSTR   = 40;   // LDS row stride (bf16 units), 80B, 16B-aligned

typedef __attribute__((ext_vector_type(16))) __bf16 v16bf;
typedef __attribute__((ext_vector_type(8)))  float  v8f;

struct U4x2 { uint4 a, b; };

static __device__ __forceinline__ __bf16 f2bf_rne(float f) {
  unsigned u = __builtin_bit_cast(unsigned, f);
  unsigned r = u + 0x7fffu + ((u >> 16) & 1u);
  return __builtin_bit_cast(__bf16, (unsigned short)(r >> 16));
}
static __device__ __forceinline__ __bf16 f2bf_trunc(float f) {
  unsigned u = __builtin_bit_cast(unsigned, f);
  return __builtin_bit_cast(__bf16, (unsigned short)(u >> 16));
}
static __device__ __forceinline__ float bf_trunc_rem(float f) {
  unsigned uh = __builtin_bit_cast(unsigned, f) & 0xffff0000u;
  return f - __builtin_bit_cast(float, uh);
}
// Pack 16 bf16 (two 16B chunks) into a WMMA fragment register block.
static __device__ __forceinline__ v16bf load_frag(const __bf16* p0, const __bf16* p1) {
  U4x2 u;
  u.a = *(const uint4*)p0;
  u.b = *(const uint4*)p1;
  return __builtin_bit_cast(v16bf, u);
}

// ---------------- Phase 1: dynamic conv kernels -> bf16 hi/lo planes -------
__global__ __launch_bounds__(256) void build_kernels(
    const float* __restrict__ Q, const float* __restrict__ Wq,
    const float* __restrict__ bq,
    __bf16* __restrict__ Ahi, __bf16* __restrict__ Alo)
{
  const int tid = threadIdx.x;
  const int kw  = tid & 31;                 // lane = kw column (coalesced Wq)
  const int cl  = tid >> 5;                 // 0..7
  const int o   = blockIdx.x >> 7;          // blockIdx.x / 128
  const int c   = ((blockIdx.x & 127) << 3) + cl;

  const float* qrow = Q + ((size_t)o * SQ + c) * DM;   // broadcast across wave
  float acc = 0.f;
  for (int d = 0; d < DM; d += 4) {
    const float4 q4 = *(const float4*)(qrow + d);
    acc = fmaf(q4.x, Wq[(d + 0) * KW + kw], acc);
    acc = fmaf(q4.y, Wq[(d + 1) * KW + kw], acc);
    acc = fmaf(q4.z, Wq[(d + 2) * KW + kw], acc);
    acc = fmaf(q4.w, Wq[(d + 3) * KW + kw], acc);
  }
  acc += bq[kw];

  const int ok = o * KW + kw;
  Ahi[(size_t)ok * DM + c] = f2bf_trunc(acc);
  Alo[(size_t)ok * DM + c] = f2bf_rne(bf_trunc_rem(acc));
}

// ---------------- Phase 2: S = A (128x1024) * keys^T (1024x8192) -----------
__global__ __launch_bounds__(256) void conv_gemm_wmma(
    const float* __restrict__ keys,          // key_sequences (BN, SK, DM)
    const __bf16* __restrict__ Ahi, const __bf16* __restrict__ Alo,
    float* __restrict__ S)                   // (BN, OK, SK)
{
  __shared__ __bf16 sBh[NT_BLK * BSTR];      // keys tile, bf16 hi plane
  __shared__ __bf16 sBl[NT_BLK * BSTR];      // keys tile, bf16 lo plane

  const int tid  = threadIdx.x;
  const int lane = tid & 31;
  const int w    = tid >> 5;                 // wave id 0..7 -> M tile
  const int n    = blockIdx.y;
  const int t0   = blockIdx.x * NT_BLK;
  const int half = lane >> 4;
  const int l15  = lane & 15;

  v8f zero = {};
  v8f acc[8];
  #pragma unroll
  for (int j = 0; j < 8; ++j) acc[j] = zero;

  // cooperative staging indices: thread -> (t row, c quad)
  const int sq = tid & 7;                    // float4 slot in 32-wide c
  const int st = tid >> 3;                   // base t row 0..31
  const float* kbase = keys + ((size_t)n * SK + t0) * DM;

  // A fragment source (row = ok, contiguous in c)
  const int okRow = w * 16 + l15;
  const __bf16* ah = Ahi + (size_t)okRow * DM;
  const __bf16* al = Alo + (size_t)okRow * DM;
  const int koff = half * 8;                 // ISA 16-bit A layout per lane half

  for (int c0 = 0; c0 < DM; c0 += KSTEP) {
    // ---- stage 128x32 fp32 keys -> bf16 hi/lo planes in LDS ----
    #pragma unroll
    for (int r = 0; r < 4; ++r) {
      const int t = st + r * 32;
      const float* src = kbase + (size_t)t * DM + c0 + sq * 4;
      const float4 kv = *(const float4*)src;
      if (c0 + KSTEP < DM)                   // gfx1250 global_prefetch_b8
        __builtin_prefetch(src + KSTEP, 0, 0);
      const float f[4] = {kv.x, kv.y, kv.z, kv.w};
      #pragma unroll
      for (int i = 0; i < 4; ++i) {
        sBh[t * BSTR + sq * 4 + i] = f2bf_trunc(f[i]);
        sBl[t * BSTR + sq * 4 + i] = f2bf_rne(bf_trunc_rem(f[i]));
      }
    }
    __syncthreads();

    // A fragments: lanes 0-15 hold K 0-7 / 16-23, lanes 16-31 K 8-15 / 24-31
    const v16bf aH = load_frag(ah + c0 + koff, ah + c0 + koff + 16);
    const v16bf aL = load_frag(al + c0 + koff, al + c0 + koff + 16);

    #pragma unroll
    for (int j = 0; j < 8; ++j) {
      // B fragment: lane = column t, lanes 0-15 K 0-7 / 16-23, lanes 16-31 K 8-15 / 24-31
      const __bf16* bh = &sBh[(j * 16 + l15) * BSTR + half * 8];
      const __bf16* bl = &sBl[(j * 16 + l15) * BSTR + half * 8];
      const v16bf bH = load_frag(bh, bh + 16);
      const v16bf bL = load_frag(bl, bl + 16);
      // 3-term bf16 split: ah*bh + ah*bl + al*bh  (~fp32 accuracy)
      acc[j] = __builtin_amdgcn_wmma_f32_16x16x32_bf16(
          false, aH, false, bH, (short)0, acc[j], false, false);
      acc[j] = __builtin_amdgcn_wmma_f32_16x16x32_bf16(
          false, aH, false, bL, (short)0, acc[j], false, false);
      acc[j] = __builtin_amdgcn_wmma_f32_16x16x32_bf16(
          false, aL, false, bH, (short)0, acc[j], false, false);
    }
    __syncthreads();
  }

  // C/D layout: VGPR v -> M=v (lanes<16) / M=v+8 (lanes>=16); N = lane&15
  float* sn = S + (size_t)n * OK * SK;
  #pragma unroll
  for (int j = 0; j < 8; ++j) {
    #pragma unroll
    for (int v = 0; v < 8; ++v) {
      const int okg = w * 16 + v + half * 8;
      sn[(size_t)okg * SK + t0 + j * 16 + l15] = acc[j][v];
    }
  }
}

// ---------------- Phase 3: diagonal reduce over k --------------------------
__global__ __launch_bounds__(256) void diag_reduce(
    const float* __restrict__ S, float* __restrict__ out)
{
  const int idx = blockIdx.x * 256 + threadIdx.x;
  if (idx >= BN * BN * LOUT) return;
  const int n   = idx / (BN * LOUT);
  const int rem = idx - n * (BN * LOUT);
  const int o   = rem / LOUT;
  const int l   = rem - o * LOUT;

  const float* sp = S + (size_t)(n * OK + o * KW) * SK;
  float acc = 0.f;
  #pragma unroll
  for (int k = 0; k < KW; ++k) {
    const int t = l + k - PAD;
    if (t >= 0 && t < SK) acc += sp[(size_t)k * SK + t];
  }
  out[idx] = acc;
}

// ---------------------------------------------------------------------------
extern "C" void kernel_launch(void* const* d_in, const int* in_sizes, int n_in,
                              void* d_out, int out_size, void* d_ws, size_t ws_size,
                              hipStream_t stream) {
  const float* Q    = (const float*)d_in[0];
  const float* Kseq = (const float*)d_in[1];
  // d_in[2] (values) and d_in[3] (lengths) are unused by the reference output.
  const float* Wq = (const float*)d_in[4];
  const float* bq = (const float*)d_in[5];

  // workspace: A hi plane (256KB) | A lo plane (256KB) | S fp32 (16MB)
  __bf16* Ahi = (__bf16*)d_ws;
  __bf16* Alo = Ahi + (size_t)OK * DM;
  float*  S   = (float*)((char*)d_ws + 2 * (size_t)OK * DM * sizeof(__bf16));
  float*  out = (float*)d_out;

  build_kernels<<<BN * (SQ / 8), 256, 0, stream>>>(Q, Wq, bq, Ahi, Alo);

  dim3 g2(SK / NT_BLK, BN);
  conv_gemm_wmma<<<g2, 256, 0, stream>>>(Kseq, Ahi, Alo, S);

  const int total = BN * BN * LOUT;
  diag_reduce<<<(total + 255) / 256, 256, 0, stream>>>(S, out);
}
